// RQVAEModel_84318797955169
// MI455X (gfx1250) — compile-verified, matrix operand order checked by reading, not run on hardware
//
#include <hip/hip_runtime.h>
#include <math.h>

// ---------------- problem constants ----------------
#define NROWS   32768
#define DIN     768
#define DLAT    256
#define DEPTH   4
#define KCB     1024

// d_out element offsets (f32 elements; indices written as int32 bits)
#define OFF_RECON   0
#define OFF_IDX     (NROWS*DIN)                    // 25,165,824
#define OFF_LATENT  (OFF_IDX + DEPTH*NROWS)        // 25,296,896
#define OFF_QSUM    (OFF_LATENT + NROWS*DLAT)      // 33,685,504

// padded LDS row stride for codebook tiles (256 + 8 bf16 = 528B -> conflict-free b128 reads)
#define ROWP 264

typedef __bf16 bfv8  __attribute__((ext_vector_type(8)));
typedef __bf16 bfv16 __attribute__((ext_vector_type(16)));
typedef float  v8f   __attribute__((ext_vector_type(8)));

// Load one WMMA 16-bit operand tile chunk for this lane.
// Caller passes p = row_base_ptr + k + 8*h  (16B aligned); second chunk at +16.
static __device__ __forceinline__ bfv16 ld_tile(const __bf16* p) {
    bfv8 lo = *(const bfv8*)(p);
    bfv8 hi = *(const bfv8*)(p + 16);
    return __builtin_shufflevector(lo, hi, 0,1,2,3,4,5,6,7,8,9,10,11,12,13,14,15);
}

static __device__ __forceinline__ float gelu_exact(float x) {
    return 0.5f * x * (1.0f + erff(x * 0.70710678118654752f));
}

static __device__ __forceinline__ void sched_fence() {
#if __has_builtin(__builtin_amdgcn_sched_barrier)
    __builtin_amdgcn_sched_barrier(0);   // keep load block and compute block separated
#endif
}

// ---------------- Tensor Data Mover staging (guarded) ----------------
#if __has_builtin(__builtin_amdgcn_tensor_load_to_lds) && \
    __has_builtin(__builtin_amdgcn_s_wait_tensorcnt)
#define USE_TDM 1
#endif

#ifdef USE_TDM
typedef unsigned int u32x4 __attribute__((ext_vector_type(4)));
typedef int          i32x4 __attribute__((ext_vector_type(4)));
typedef int          i32x8 __attribute__((ext_vector_type(8)));

// DMA one codebook tile (16 rows x 256 bf16 = 8192B contiguous) into LDS at
// lds_addr, inserting 16B of pad every 512B (LDS row stride = 528B = ROWP*2).
// D# per CDNA5 ISA 8.3-8.6: count=1, type=2(image), data_size=2B,
// 1-D tile of 4096 elements, pad_interval=128 dwords (code 6), pad_amount=4 dwords (code 3).
static __device__ __forceinline__ void tdm_load_tile(const void* gsrc, unsigned lds_addr) {
    unsigned long long ga = (unsigned long long)(uintptr_t)gsrc;
    u32x4 g0 = { 1u,                                    // count=1 (valid descriptor)
                 lds_addr,                              // lds_addr (bytes)
                 (unsigned)ga,                          // global_addr[31:0]
                 ((unsigned)((ga >> 32) & 0x01FFFFFFull)) | 0x80000000u }; // [56:32] | type=2
    i32x8 g1 = { (int)((1u << 16)                      // data_size = 2 bytes
                     | (1u << 20)                      // pad_enable
                     | (6u << 22)                      // pad_interval: 128 dwords
                     | (3u << 25)),                    // pad_amount: 4 dwords
                 (int)(4096u << 16),                   // tensor_dim0[15:0] = 4096 @ bits[31:16]
                 (int)(1u << 16),                      // tensor_dim0 hi=0 | tensor_dim1 lo=1
                 (int)(4096u << 16),                   // tensor_dim1 hi=0 | tile_dim0 = 4096
                 0,                                    // tile_dim1 = 0, tile_dim2 = 0
                 4096,                                 // tensor_dim0_stride lo (unused, sane)
                 0, 0 };
    i32x4 gz4 = { 0, 0, 0, 0 };
#if __clang_major__ >= 23
    i32x8 gz8 = { 0, 0, 0, 0, 0, 0, 0, 0 };
    __builtin_amdgcn_tensor_load_to_lds(g0, g1, gz4, gz4, gz8, 0);
#else
    __builtin_amdgcn_tensor_load_to_lds(g0, g1, gz4, gz4, 0);
#endif
}
#endif // USE_TDM

// issue a tile stage: gsrc = 16x256 bf16 contiguous; lbuf = LDS tile (16 rows, stride ROWP)
static __device__ __forceinline__ void stage_issue(const __bf16* gsrc, __bf16* lbuf, int tid) {
#ifdef USE_TDM
    if (tid < 32) {   // one wave issues the DMA (wave-uniform branch; TDM ignores EXEC)
        tdm_load_tile((const void*)gsrc, (unsigned)(uintptr_t)lbuf);
    }
#else
    // cooperative copy: 256 threads x 32B
    int r = tid >> 4;
    int c = (tid & 15) * 16;
    const uint4* s = (const uint4*)(gsrc + r * 256 + c);
    uint4* d = (uint4*)(lbuf + r * ROWP + c);
    d[0] = s[0];
    d[1] = s[1];
#endif
}

static __device__ __forceinline__ void stage_wait(int tid) {
#ifdef USE_TDM
    if (tid < 32) __builtin_amdgcn_s_wait_tensorcnt(0);
#endif
}

// ---------------- prep kernels ----------------
__global__ void k_cvt_bf16(const float* __restrict__ src, __bf16* __restrict__ dst, int n) {
    int i = blockIdx.x * blockDim.x + threadIdx.x;
    if (i < n) dst[i] = (__bf16)src[i];
}

// W[K,N] (f32, row-major) -> Wt[N,K] (bf16, row-major)
__global__ void k_transpose_bf16(const float* __restrict__ W, __bf16* __restrict__ Wt,
                                 int K, int N) {
    int i = blockIdx.x * blockDim.x + threadIdx.x;
    if (i < K * N) {
        int n = i / K, k = i % K;
        Wt[i] = (__bf16)W[(size_t)k * N + n];
    }
}

// squared L2 norm of each codebook row (DEPTH*KCB rows of DLAT f32)
__global__ void k_cbnorm(const float* __restrict__ cb, float* __restrict__ norms) {
    int j = blockIdx.x * blockDim.x + threadIdx.x;
    if (j < DEPTH * KCB) {
        const float* r = cb + (size_t)j * DLAT;
        float s = 0.f;
        #pragma unroll 8
        for (int c = 0; c < DLAT; ++c) { float v = r[c]; s += v * v; }
        norms[j] = s;
    }
}

__global__ void k_zero_f32(float* __restrict__ p, int n) {
    int i = blockIdx.x * blockDim.x + threadIdx.x;
    if (i < n) p[i] = 0.f;
}

// ---------------- WMMA GEMM: C[M,N] = act(A[M,K](bf16) @ Wt[N,K]^T + bias) ----------------
// block: 256 threads = 8 waves, arranged 2(M)x4(N); wave tile 16x64; block tile 32x256.
// K, N compile-time: K loop fully unrolls -> software pipeline resolves via SSA renaming
// (no register moves) and all addresses fold into immediate offsets.
// grid: (N/256, M/32)
template<int K, int N, int DO_GELU>
__global__ void __launch_bounds__(256)
k_gemm(const __bf16* __restrict__ A, const __bf16* __restrict__ Wt,
       const float* __restrict__ bias,
       float* __restrict__ Cf, float* __restrict__ Cf2, __bf16* __restrict__ Cb) {
    const int lane = threadIdx.x & 31;
    const int wv   = threadIdx.x >> 5;
    const int wm   = wv >> 2, wn = wv & 3;
    const int lrow = lane & 15, h = lane >> 4;

    const int rowbase = blockIdx.y * 32 + wm * 16;
    const int colbase = blockIdx.x * 256 + wn * 64;

    v8f acc[4];
    #pragma unroll
    for (int t = 0; t < 4; ++t) acc[t] = (v8f){0.f,0.f,0.f,0.f,0.f,0.f,0.f,0.f};

    const __bf16* arow = A + (size_t)(rowbase + lrow) * K + 8 * h;
    const __bf16* brow[4];
    #pragma unroll
    for (int t = 0; t < 4; ++t)
        brow[t] = Wt + (size_t)(colbase + t * 16 + lrow) * K + 8 * h;

    // prologue: fragments for k = 0
    bfv16 a = ld_tile(arow);
    bfv16 b[4];
    #pragma unroll
    for (int t = 0; t < 4; ++t) b[t] = ld_tile(brow[t]);

    #pragma unroll
    for (int k = 32; k < K; k += 32) {
        // prefetch next fragments before consuming current ones
        bfv16 an = ld_tile(arow + k);
        bfv16 bn[4];
        #pragma unroll
        for (int t = 0; t < 4; ++t) bn[t] = ld_tile(brow[t] + k);

        #pragma unroll
        for (int t = 0; t < 4; ++t)
            acc[t] = __builtin_amdgcn_wmma_f32_16x16x32_bf16(
                false, a, false, b[t], (short)0, acc[t], false, false);

        a = an;
        #pragma unroll
        for (int t = 0; t < 4; ++t) b[t] = bn[t];
    }
    #pragma unroll
    for (int t = 0; t < 4; ++t)
        acc[t] = __builtin_amdgcn_wmma_f32_16x16x32_bf16(
            false, a, false, b[t], (short)0, acc[t], false, false);

    #pragma unroll
    for (int t = 0; t < 4; ++t) {
        #pragma unroll
        for (int v = 0; v < 8; ++v) {
            int row = rowbase + v + 8 * h;
            int col = colbase + t * 16 + lrow;
            float val = acc[t][v] + bias[col];
            if (DO_GELU) val = gelu_exact(val);
            size_t o = (size_t)row * N + col;
            if (Cf)  Cf[o]  = val;
            if (Cf2) Cf2[o] = val;
            if (Cb)  Cb[o]  = (__bf16)val;
        }
    }
}

// ---------------- fused RQ step (one depth) ----------------
// WMMA scores (residual strip held in registers, codebook tiles TDM-staged in LDS,
// double-buffered) + wave argmin reduction + cooperative residual/qsum update.
// All 16 ds_load_b128 for a tile are issued as one block (8 live fragments) before
// the WMMA chain, fenced with sched_barrier, so dscnt waits are partial and LDS
// latency hides under earlier WMMAs.
// block: 256 threads = 8 waves; wave owns 16 rows -> block owns 128 rows. grid: NROWS/128.
__global__ void __launch_bounds__(256)
k_rq_step(const __bf16* __restrict__ cb_b,    // this depth's bf16 codebook [1024,256]
          const float*  __restrict__ cb_f,    // this depth's f32 codebook  [1024,256]
          const float*  __restrict__ cbnorm,  // this depth's ||c||^2 [1024]
          float* __restrict__ Rf,             // residual f32 [N,256] (in/out)
          __bf16* __restrict__ Rb,            // residual bf16 [N,256] (in/out)
          float* __restrict__ Qsum,           // qsum f32 (d_out) [N,256] (in/out)
          __bf16* __restrict__ Qb,            // qsum bf16 [N,256] (out)
          int* __restrict__ idx_out)          // indices for this depth [N]
{
    __shared__ __align__(16) __bf16 s_buf[2][16 * ROWP];
    __shared__ int s_idx[128];

    const int tid  = threadIdx.x;
    const int lane = tid & 31;
    const int wv   = tid >> 5;
    const int lrow = lane & 15, h = lane >> 4;
    const int rbase = blockIdx.x * 128 + wv * 16;

    // hoist the wave's 16x256 residual strip into registers (loop-invariant A operand)
    bfv16 aK[8];
    {
        const __bf16* arow = Rb + (size_t)(rbase + lrow) * DLAT + 8 * h;
        #pragma unroll
        for (int kk = 0; kk < 8; ++kk) aK[kk] = ld_tile(arow + kk * 32);
    }

    // stage tile 0
    stage_issue(cb_b, s_buf[0], tid);
    stage_wait(tid);
    __syncthreads();

    float mv[8]; int mi[8];
    #pragma unroll
    for (int v = 0; v < 8; ++v) { mv[v] = 3.4e38f; mi[v] = 0; }

    for (int t = 0; t < KCB / 16; ++t) {
        if (t + 1 < KCB / 16)
            stage_issue(cb_b + (size_t)(t + 1) * 16 * DLAT, s_buf[(t + 1) & 1], tid);

        const __bf16* bbase = &s_buf[t & 1][lrow * ROWP + 8 * h];

        // issue the whole tile's LDS reads first: 8 fragments simultaneously live,
        // so the allocator cannot serialize them into one register pair.
        bfv16 bf[8];
        #pragma unroll
        for (int kk = 0; kk < 8; ++kk) bf[kk] = ld_tile(bbase + kk * 32);

        sched_fence();   // keep the load block above the WMMA chain

        v8f c = (v8f){0.f,0.f,0.f,0.f,0.f,0.f,0.f,0.f};
        #pragma unroll
        for (int kk = 0; kk < 8; ++kk)
            c = __builtin_amdgcn_wmma_f32_16x16x32_bf16(
                false, aK[kk], false, bf[kk], (short)0, c, false, false);

        int j = t * 16 + lrow;           // code index seen by this lane (all 8 rows)
        float cn = cbnorm[j];
        #pragma unroll
        for (int v = 0; v < 8; ++v) {
            float d = cn - 2.0f * c[v];  // ||r||^2 term is row-constant: argmin-invariant
            if (d < mv[v]) { mv[v] = d; mi[v] = j; }
        }

        stage_wait(tid);                 // next tile landed in LDS
        __syncthreads();                 // ...and everyone is done with the current buffer
    }

    // reduce min across the 16 lanes of each half (rows are fixed per (v,half))
    #pragma unroll
    for (int v = 0; v < 8; ++v) {
        float bv = mv[v]; int bi = mi[v];
        #pragma unroll
        for (int off = 8; off >= 1; off >>= 1) {
            float ov = __shfl_xor(bv, off, 32);
            int   oi = __shfl_xor(bi, off, 32);
            if (ov < bv || (ov == bv && oi < bi)) { bv = ov; bi = oi; }
        }
        if (lrow == 0) {
            int rl = v + 8 * h;                      // local row 0..15 in this wave's strip
            s_idx[wv * 16 + rl] = bi;
            idx_out[rbase + rl] = bi;
        }
    }
    __syncthreads();

    // cooperative gather + residual/qsum update over the block's 128x256 slab
    for (int i = tid; i < 128 * DLAT; i += 256) {
        int r = i >> 8, col = i & (DLAT - 1);
        int grow = blockIdx.x * 128 + r;
        int j = s_idx[r];
        float q  = cb_f[(size_t)j * DLAT + col];
        size_t o = (size_t)grow * DLAT + col;
        float nr = Rf[o] - q;
        Rf[o] = nr;
        Rb[o] = (__bf16)nr;
        float qs = Qsum[o] + q;
        Qsum[o] = qs;
        Qb[o]   = (__bf16)qs;
    }
}

// ---------------- launcher ----------------
extern "C" void kernel_launch(void* const* d_in, const int* in_sizes, int n_in,
                              void* d_out, int out_size, void* d_ws, size_t ws_size,
                              hipStream_t stream) {
    (void)in_sizes; (void)n_in; (void)out_size; (void)ws_size;

    const float* inputs  = (const float*)d_in[0];
    const float* enc_w1  = (const float*)d_in[1];
    const float* enc_b1  = (const float*)d_in[2];
    const float* enc_w2  = (const float*)d_in[3];
    const float* enc_b2  = (const float*)d_in[4];
    const float* cb      = (const float*)d_in[5];
    const float* dec_w1  = (const float*)d_in[6];
    const float* dec_b1  = (const float*)d_in[7];
    const float* dec_w2  = (const float*)d_in[8];
    const float* dec_b2  = (const float*)d_in[9];

    float* out    = (float*)d_out;
    float* recon  = out + OFF_RECON;
    int*   idxo   = (int*)(out + OFF_IDX);
    float* latent = out + OFF_LATENT;
    float* qsum   = out + OFF_QSUM;

    // ---- workspace carve-up (bytes) ----
    char* ws = (char*)d_ws;
    size_t o = 0;
    __bf16* in_b   = (__bf16*)(ws + o); o += (size_t)NROWS * DIN * 2;   // 48 MB (reused as h2)
    __bf16* h1_b   = (__bf16*)(ws + o); o += (size_t)NROWS * DLAT * 2;  // 16 MB
    float*  res_f  = (float*) (ws + o); o += (size_t)NROWS * DLAT * 4;  // 32 MB
    __bf16* res_b  = (__bf16*)(ws + o); o += (size_t)NROWS * DLAT * 2;  // 16 MB
    __bf16* qs_b   = (__bf16*)(ws + o); o += (size_t)NROWS * DLAT * 2;  // 16 MB
    __bf16* ew1t   = (__bf16*)(ws + o); o += (size_t)DLAT * DIN * 2;
    __bf16* ew2t   = (__bf16*)(ws + o); o += (size_t)DLAT * DLAT * 2;
    __bf16* dw1t   = (__bf16*)(ws + o); o += (size_t)DLAT * DLAT * 2;
    __bf16* dw2t   = (__bf16*)(ws + o); o += (size_t)DIN * DLAT * 2;
    __bf16* cb_b   = (__bf16*)(ws + o); o += (size_t)DEPTH * KCB * DLAT * 2;
    float*  cbn    = (float*) (ws + o); o += (size_t)DEPTH * KCB * 4;
    __bf16* h2_b   = in_b;  // alias: inputs_bf16 dead after encoder GEMM1

    // ---- prep ----
    {
        int n = NROWS * DIN;
        k_cvt_bf16<<<(n + 255) / 256, 256, 0, stream>>>(inputs, in_b, n);
    }
    {
        int n = DEPTH * KCB * DLAT;
        k_cvt_bf16<<<(n + 255) / 256, 256, 0, stream>>>(cb, cb_b, n);
    }
    k_transpose_bf16<<<(DIN * DLAT + 255) / 256, 256, 0, stream>>>(enc_w1, ew1t, DIN, DLAT);
    k_transpose_bf16<<<(DLAT * DLAT + 255) / 256, 256, 0, stream>>>(enc_w2, ew2t, DLAT, DLAT);
    k_transpose_bf16<<<(DLAT * DLAT + 255) / 256, 256, 0, stream>>>(dec_w1, dw1t, DLAT, DLAT);
    k_transpose_bf16<<<(DLAT * DIN + 255) / 256, 256, 0, stream>>>(dec_w2, dw2t, DLAT, DIN);
    k_cbnorm<<<(DEPTH * KCB + 255) / 256, 256, 0, stream>>>(cb, cbn);
    k_zero_f32<<<((NROWS * DLAT) + 255) / 256, 256, 0, stream>>>(qsum, NROWS * DLAT);

    // ---- encoder ----
    // h1 = GELU(X @ enc_w1 + b1) : M=32768, N=256, K=768
    k_gemm<DIN, DLAT, 1><<<dim3(DLAT / 256, NROWS / 32), 256, 0, stream>>>(
        in_b, ew1t, enc_b1, nullptr, nullptr, h1_b);
    // latent = h1 @ enc_w2 + b2 ; also seeds residual (f32 + bf16)
    k_gemm<DLAT, DLAT, 0><<<dim3(DLAT / 256, NROWS / 32), 256, 0, stream>>>(
        h1_b, ew2t, enc_b2, latent, res_f, res_b);

    // ---- residual quantization, 4 depths ----
    for (int d = 0; d < DEPTH; ++d) {
        k_rq_step<<<NROWS / 128, 256, 0, stream>>>(
            cb_b + (size_t)d * KCB * DLAT,
            cb   + (size_t)d * KCB * DLAT,
            cbn  + (size_t)d * KCB,
            res_f, res_b, qsum, qs_b,
            idxo + (size_t)d * NROWS);
    }

    // ---- decoder ----
    // h2 = GELU(qsum @ dec_w1 + b1) : M=32768, N=256, K=256
    k_gemm<DLAT, DLAT, 1><<<dim3(DLAT / 256, NROWS / 32), 256, 0, stream>>>(
        qs_b, dw1t, dec_b1, nullptr, nullptr, h2_b);
    // recon = h2 @ dec_w2 + b2 : M=32768, N=768, K=256
    k_gemm<DLAT, DIN, 0><<<dim3(DIN / 256, NROWS / 32), 256, 0, stream>>>(
        h2_b, dw2t, dec_b2, recon, nullptr, nullptr);
}
